// VelocityExtractor_26207890440218
// MI455X (gfx1250) — compile-verified, matrix operand order checked by reading, not run on hardware
//
#include <hip/hip_runtime.h>
#include <hip/hip_bf16.h>

typedef __attribute__((ext_vector_type(16))) _Float16 v16h;
typedef __attribute__((ext_vector_type(8)))  float    v8f;

#define OUT_DIM 224
#define NBINS   8
#define THREADS 256

__global__ __launch_bounds__(THREADS)
void velhist_kernel(const float* __restrict__ flows,
                    const float* __restrict__ boxes,
                    float* __restrict__ out,
                    int H, int W)
{
    // rows 0..7: per-thread bin sums, rows 8..15: per-thread bin counts.
    // Slot (row*256 + tid): bank = tid%64 -> ds_add_f32 is always conflict-free.
    __shared__ float hist[16 * THREADS];
    // Per-axis precomputed sampling state: {lo, hi, frac, validMul} packed in int4.
    // X entries hold column indices; Y entries hold row offsets pre-multiplied by W.
    __shared__ int4 Xc[OUT_DIM];
    __shared__ int4 Yc[OUT_DIM];

    const int tid = threadIdx.x;
    const int m   = blockIdx.x;

    #pragma unroll
    for (int r = 0; r < 16; ++r) hist[r * THREADS + tid] = 0.0f;

    const float bidxf = boxes[m * 5 + 0];
    const float bx1   = boxes[m * 5 + 1];
    const float by1   = boxes[m * 5 + 2];
    const float bx2   = boxes[m * 5 + 3];
    const float by2   = boxes[m * 5 + 4];
    const int   bidx  = (int)bidxf;
    const float sx = fmaxf(bx2 - bx1, 1.0f) * (1.0f / OUT_DIM);
    const float sy = fmaxf(by2 - by1, 1.0f) * (1.0f / OUT_DIM);

    // Hoist all per-axis sampling math out of the 50176-sample loop.
    for (int t = tid; t < OUT_DIM; t += THREADS) {
        const float px  = bx1 + ((float)t + 0.5f) * sx;
        const float vx  = ((px >= -1.0f) && (px <= (float)W)) ? 1.0f : 0.0f;
        const float pcx = fminf(fmaxf(px, 0.0f), (float)(W - 1));
        const int   x0  = (int)pcx;                 // pcx >= 0, floor == trunc
        const int   x1i = min(x0 + 1, W - 1);
        const float fx  = pcx - (float)x0;
        Xc[t] = make_int4(x0, x1i, __float_as_int(fx), __float_as_int(vx));

        const float py  = by1 + ((float)t + 0.5f) * sy;
        const float vy  = ((py >= -1.0f) && (py <= (float)H)) ? 1.0f : 0.0f;
        const float pcy = fminf(fmaxf(py, 0.0f), (float)(H - 1));
        const int   y0  = (int)pcy;
        const int   y1i = min(y0 + 1, H - 1);
        const float fy  = pcy - (float)y0;
        Yc[t] = make_int4(y0 * W, y1i * W, __float_as_int(fy), __float_as_int(vy));
    }
    __syncthreads();

    const int HW = H * W;
    const float* __restrict__ p0 = flows + (long long)bidx * 2 * HW; // channel 0
    const float* __restrict__ p1 = p0 + HW;                          // channel 1

    // 224*224 = 50176 = 196 * 256 -> uniform trip count, EXEC stays full.
    for (int s = tid; s < OUT_DIM * OUT_DIM; s += THREADS) {
        const unsigned us = (unsigned)s;
        const int j = (int)(us / OUT_DIM);
        const int i = (int)(us - (unsigned)j * OUT_DIM);

        const int4 xc = Xc[i];              // ds_load_b128
        const int4 yc = Yc[j];              // ds_load_b128
        const float fx = __int_as_float(xc.z);
        const float fy = __int_as_float(yc.z);
        const float vm = __int_as_float(xc.w) * __int_as_float(yc.w); // 1 or 0

        const float gx0 = 1.0f - fx, gy0 = 1.0f - fy;
        const float w00 = gy0 * gx0, w01 = gy0 * fx;
        const float w10 = fy  * gx0, w11 = fy  * fx;

        const int o00 = yc.x + xc.x;        // 32-bit offsets: SGPR base + VGPR off
        const int o01 = yc.x + xc.y;
        const int o10 = yc.y + xc.x;
        const int o11 = yc.y + xc.y;

        float c0 = p0[o00] * w00 + p0[o01] * w01 + p0[o10] * w10 + p0[o11] * w11;
        float c1 = p1[o00] * w00 + p1[o01] * w01 + p1[o10] * w10 + p1[o11] * w11;
        c0 *= vm;
        c1 *= vm;

        const float mag = sqrtf(c0 * c0 + c1 * c1);

        // Octant binning: bin = floor((atan2(c0,c1)+pi)*4/pi) without atan2.
        // Boundaries at multiples of pi/4 depend only on signs and |c0| vs |c1|.
        const float ay = fabsf(c0), ax = fabsf(c1);
        int bin = (c0 >= 0.0f)
                    ? ((c1 > 0.0f) ? ((ay < ax) ? 4 : 5) : ((ay > ax) ? 6 : 7))
                    : ((c1 < 0.0f) ? ((ay < ax) ? 0 : 1) : ((ay > ax) ? 2 : 3));
        bin = ((c0 == 0.0f) && (c1 == 0.0f)) ? 4 : bin;  // atan2(0,0)=0 -> bin 4 (masked samples)

        const int slot = bin * THREADS + tid;
        atomicAdd(&hist[slot], mag);                 // ds_add_f32, conflict-free
        atomicAdd(&hist[slot + 8 * THREADS], 1.0f);  // ds_add_f32, conflict-free
    }
    __syncthreads();

    // Column-sum reduction of hist[16][256] on the matrix pipe:
    // D = ones(16x32) * B(32x16) accumulated over 8 chunks of 32.
    // Columns 0..7 = bin sums, 8..15 = bin counts. A==1 makes the exact
    // K-slot mapping irrelevant: lanes L and L+16 jointly cover the 32
    // chunk entries of column L%16. f16 hi/lo split keeps sums ~f32-exact;
    // counts (<=196 per thread) are exact integers in f16.
    if (tid < 32) {
        const int lane  = tid;            // wave32: tid == lane for wave 0
        const int n     = lane & 15;      // column owned by this lane
        const int khalf = (lane & 16) ? 16 : 0;

        v16h ones;
        #pragma unroll
        for (int q = 0; q < 16; ++q) ones[q] = (_Float16)1.0f;

        v8f acc = {};
        #pragma unroll
        for (int c = 0; c < 8; ++c) {
            v16h bhi, blo;
            #pragma unroll
            for (int q = 0; q < 16; ++q) {
                const float v = hist[n * THREADS + c * 32 + khalf + q];
                const _Float16 h = (_Float16)v;
                bhi[q] = h;
                blo[q] = (_Float16)(v - (float)h);
            }
            acc = __builtin_amdgcn_wmma_f32_16x16x32_f16(
                      false, ones, false, bhi, (short)0, acc, false, false);
            acc = __builtin_amdgcn_wmma_f32_16x16x32_f16(
                      false, ones, false, blo, (short)0, acc, false, false);
        }

        // All D rows identical; VGPR0 lane n holds column-n total.
        const float v0  = acc[0];
        const float cnt = __shfl(v0, lane + 8, 32); // lane b pulls count from lane b+8
        if (lane < NBINS) {
            out[m * NBINS + lane] = (cnt != 0.0f) ? (v0 / fmaxf(cnt, 1.0f)) : 0.0f;
        }
    }
}

extern "C" void kernel_launch(void* const* d_in, const int* in_sizes, int n_in,
                              void* d_out, int out_size, void* d_ws, size_t ws_size,
                              hipStream_t stream) {
    const float* flows = (const float*)d_in[0];
    const float* boxes = (const float*)d_in[1];
    float* out = (float*)d_out;

    const int M = in_sizes[1] / 5;   // boxes: (M, 5)
    const int H = 512, W = 512;      // reference setup: flows (8, 2, 512, 512)

    velhist_kernel<<<dim3(M), dim3(THREADS), 0, stream>>>(flows, boxes, out, H, W);
}
